// AmbienceTransformer_80384607912397
// MI455X (gfx1250) — matrix operand order repaired and checked by reading, not compiled, not run on hardware
//
#include <hip/hip_runtime.h>
#include <hip/hip_bf16.h>
#include <math.h>

// MI455X / gfx1250, wave32. All matmuls via v_wmma_f32_16x16x32_f16.
// f16 operand flow end-to-end (weights pre-transposed+converted once per
// launch); f32 residual stream. GEMM tiles staged by the Tensor Data Mover
// (tensor_load_to_lds, TENSORcnt) when available; attention staging uses
// async global->LDS copies (ASYNCcnt). Fallbacks keep the file compiling on
// toolchains without those builtins.

typedef __attribute__((ext_vector_type(16))) _Float16 v16h;
typedef __attribute__((ext_vector_type(8)))  _Float16 v8h;
typedef __attribute__((ext_vector_type(4)))  _Float16 v4h;
typedef __attribute__((ext_vector_type(8)))  float    v8f;
typedef __attribute__((ext_vector_type(4)))  float    vf4;
typedef __attribute__((ext_vector_type(4)))  int      v4i;
typedef __attribute__((ext_vector_type(8)))  int      v8i;
typedef __attribute__((ext_vector_type(4)))  unsigned int v4u;

union V16 { v16h v; v8h h[2]; };

#if defined(__HIP_DEVICE_COMPILE__) && \
    __has_builtin(__builtin_amdgcn_global_load_async_to_lds_b128)
#define HAS_ASYNC_CP 1
typedef __attribute__((address_space(1))) v4i* gas_p;   // global int4*
typedef __attribute__((address_space(3))) v4i* las_p;   // LDS int4*
#else
#define HAS_ASYNC_CP 0
#endif

#if defined(__HIP_DEVICE_COMPILE__) && \
    __has_builtin(__builtin_amdgcn_tensor_load_to_lds) && \
    __has_builtin(__builtin_amdgcn_s_wait_tensorcnt)
#define HAS_TDM 1
#else
#define HAS_TDM 0
#endif

__device__ __forceinline__ void async_wait0() {
#if HAS_ASYNC_CP
#if __has_builtin(__builtin_amdgcn_s_wait_asynccnt)
  __builtin_amdgcn_s_wait_asynccnt(0);
#else
  asm volatile("s_wait_asynccnt 0x0" ::: "memory");
#endif
#endif
}

// 16-byte global -> LDS copy (async DMA when available, tracked copy else).
__device__ __forceinline__ void cp16(const _Float16* g, _Float16* l) {
#if HAS_ASYNC_CP
  __builtin_amdgcn_global_load_async_to_lds_b128((gas_p)(void*)g,
                                                 (las_p)(void*)l, 0, 0);
#else
  *(v8h*)l = *(const v8h*)g;
#endif
}

#if HAS_TDM
__device__ __forceinline__ unsigned lds_addr_of(void* p) {
  return (unsigned)(size_t)(__attribute__((address_space(3))) void*)p;
}

// TDM 2D tile load: rows x rowHalves f16 tile, row stride strideHalves,
// global -> LDS (contiguous rows of rowHalves*2 bytes). D# per ISA 08 §8.
__device__ __forceinline__ void tdm_load_2d(const _Float16* gsrc,
                                            unsigned ldsByteAddr,
                                            unsigned rows, unsigned rowHalves,
                                            unsigned long long strideHalves) {
  const unsigned long long ga = (unsigned long long)(size_t)gsrc;
  v4u g0;
  g0[0] = 1u;                                          // count=1, user desc
  g0[1] = ldsByteAddr;                                 // lds_addr [63:32]
  g0[2] = (unsigned)ga;                                // global_addr lo
  g0[3] = (unsigned)((ga >> 32) & 0x01FFFFFFu) | (2u << 30);  // hi + type=2
  const unsigned td0 = rowHalves;                      // tensor_dim0 == tile
  const unsigned td1 = rows;                           // tensor_dim1 == tile
  v8i g1;
  g1[0] = (int)(1u << 16);                             // data_size=2B, mask=0
  g1[1] = (int)((td0 & 0xFFFFu) << 16);                // dim0 lo16 @ [63:48]
  g1[2] = (int)(((td0 >> 16) & 0xFFFFu) | ((td1 & 0xFFFFu) << 16));
  g1[3] = (int)(((td1 >> 16) & 0xFFFFu) | ((rowHalves & 0xFFFFu) << 16));
  g1[4] = (int)(rows & 0xFFFFu);                       // tile_dim1; tile_dim2=0
  g1[5] = (int)(unsigned)(strideHalves & 0xFFFFFFFFu); // dim0_stride lo32
  g1[6] = (int)(unsigned)((strideHalves >> 32) & 0xFFFFu);
  g1[7] = 0;                                           // dim1_stride (unused)
  v4i g2 = { 1, 1, 0, 0 };                             // dim2=1, dim3=1
  v4i g3 = { 0, (int)(1u << 16), 0, 0 };               // dim4=1
  v8i g4 = { 0, 0, 0, 0, 0, 0, 0, 0 };                 // reserved/extended
  __builtin_amdgcn_tensor_load_to_lds(g0, g1, g2, g3, g4, 0);
}
#endif

__device__ __forceinline__ v8f wmma16(v16h a, v16h b, v8f c) {
  return __builtin_amdgcn_wmma_f32_16x16x32_f16(false, a, false, b, (short)0, c,
                                                false, false);
}

__device__ __forceinline__ float gelu_f(float v) {
  return 0.5f * v * (1.0f + erff(v * 0.70710678118654752f));
}

// ---------------------------------------------------------------------------
// Transpose + convert: in[K,N] f32 -> out[N,K] f16. 64x64 tiles, block 256.
// ---------------------------------------------------------------------------
__global__ __launch_bounds__(256)
void convT_k(const float* __restrict__ in, _Float16* __restrict__ out,
             int K, int N) {
  __shared__ float t[64][65];
  const int rx = threadIdx.x & 63;
  const int ry = threadIdx.x >> 6;          // 0..3
  const int k0 = blockIdx.y * 64;
  const int n0 = blockIdx.x * 64;
#pragma unroll
  for (int i = 0; i < 16; ++i) {
    const int kk = ry * 16 + i;
    t[kk][rx] = in[(size_t)(k0 + kk) * N + n0 + rx];
  }
  __syncthreads();
#pragma unroll
  for (int i = 0; i < 16; ++i) {
    const int nn = ry * 16 + i;
    out[(size_t)(n0 + nn) * K + k0 + rx] = (_Float16)t[rx][nn];
  }
}

// ---------------------------------------------------------------------------
// Embedding gather -> f16: xh[row] = (f16)emb[tok[row]], block 256.
// ---------------------------------------------------------------------------
__global__ __launch_bounds__(256)
void embed_k(const int* __restrict__ tok, const float* __restrict__ emb,
             _Float16* __restrict__ xh) {
  const size_t row = blockIdx.x;
  const int t = tok[row];
  vf4 v = ((const vf4*)(emb + (size_t)t * 1024))[threadIdx.x];
  v4h o;
#pragma unroll
  for (int i = 0; i < 4; i++) o[i] = (_Float16)v[i];
  ((v4h*)(xh + row * 1024))[threadIdx.x] = o;
}

// ---------------------------------------------------------------------------
// f32 -> f16 elementwise (residual stream copy for the LM head A operand).
// ---------------------------------------------------------------------------
__global__ __launch_bounds__(256)
void cast_h_k(const float* __restrict__ in, _Float16* __restrict__ out) {
  const size_t i = (size_t)blockIdx.x * 256 + threadIdx.x;
  vf4 v = ((const vf4*)in)[i];
  v4h o;
#pragma unroll
  for (int j = 0; j < 4; j++) o[j] = (_Float16)v[j];
  ((v4h*)out)[i] = o;
}

// ---------------------------------------------------------------------------
// LayerNorm (D=1024): f32 in, f16 out. One row per block, 256 threads.
// ---------------------------------------------------------------------------
__global__ __launch_bounds__(256)
void ln_k(const float* __restrict__ x, const float* __restrict__ g,
          const float* __restrict__ beta, _Float16* __restrict__ o) {
  __shared__ float sred[2][8];
  __shared__ float sstat[2];
  const int tid = threadIdx.x;
  const size_t row = blockIdx.x;
  vf4 v = ((const vf4*)(x + row * 1024))[tid];
  float s = v[0] + v[1] + v[2] + v[3];
  float q = v[0]*v[0] + v[1]*v[1] + v[2]*v[2] + v[3]*v[3];
#pragma unroll
  for (int msk = 16; msk >= 1; msk >>= 1) {
    s += __shfl_xor(s, msk, 32);
    q += __shfl_xor(q, msk, 32);
  }
  if ((tid & 31) == 0) { sred[0][tid >> 5] = s; sred[1][tid >> 5] = q; }
  __syncthreads();
  if (tid == 0) {
    float ts = 0.f, tq = 0.f;
#pragma unroll
    for (int i = 0; i < 8; i++) { ts += sred[0][i]; tq += sred[1][i]; }
    const float mean = ts * (1.0f / 1024.0f);
    const float var  = tq * (1.0f / 1024.0f) - mean * mean;
    sstat[0] = mean;
    sstat[1] = rsqrtf(var + 1e-5f);
  }
  __syncthreads();
  const float mean = sstat[0], rstd = sstat[1];
  vf4 gv = ((const vf4*)g)[tid];
  vf4 bv = ((const vf4*)beta)[tid];
  v4h ov;
#pragma unroll
  for (int i = 0; i < 4; i++)
    ov[i] = (_Float16)((v[i] - mean) * rstd * gv[i] + bv[i]);
  ((v4h*)(o + row * 1024))[tid] = ov;
}

// ---------------------------------------------------------------------------
// WMMA GEMM: C[M,N] = epi(A[M,K] @ BT[N,K]^T + bias [+ res]).
// A f16 row-major, BT f16 = B transposed (both tiles are clean 2D tiles).
// Block 128x256, BK=32, 256 thr = 8 waves (2x4); wave 64x64 = 16 WMMA/K-step.
// Tiles staged by TDM (one wave issues tensor_load_to_lds) when available,
// else async per-lane copies. M%128==0, N%256==0, K%32==0 (true here).
// ---------------------------------------------------------------------------
template <bool GELU, bool RES, bool OUTH>
__global__ __launch_bounds__(256)
void gemm_h(const _Float16* __restrict__ A, const _Float16* __restrict__ BT,
            const float* __restrict__ bias, const float* __restrict__ res,
            float* __restrict__ Cf, _Float16* __restrict__ Ch,
            int M, int N, int K) {
  __shared__ _Float16 sA[2][128][32];   // [m][k]
  __shared__ _Float16 sB[2][256][32];   // [n][k]

  const int tid  = threadIdx.x;
  const int lane = tid & 31;
  const int lrow = lane & 15;
  const int lhi  = lane >> 4;
  const int w    = tid >> 5;
  const int wrow = (w >> 2) * 64;       // 0,64
  const int wcol = (w & 3) * 64;        // 0,64,128,192
  const int bm   = blockIdx.y * 128;
  const int bn   = blockIdx.x * 256;

  const v8f vzero = {};
  v8f acc[4][4];
#pragma unroll
  for (int i = 0; i < 4; i++)
#pragma unroll
    for (int j = 0; j < 4; j++) acc[i][j] = vzero;

  const int nk = K >> 5;

#if HAS_TDM
  const unsigned sAaddr = lds_addr_of(&sA[0][0][0]);
  const unsigned sBaddr = lds_addr_of(&sB[0][0][0]);
  auto stage = [&](int kt, int buf) {
    if (w == 0) {
      tdm_load_2d(A + (size_t)bm * K + (size_t)kt * 32,
                  sAaddr + (unsigned)buf * (128u * 32u * 2u), 128, 32,
                  (unsigned long long)K);
      tdm_load_2d(BT + (size_t)bn * K + (size_t)kt * 32,
                  sBaddr + (unsigned)buf * (256u * 32u * 2u), 256, 32,
                  (unsigned long long)K);
    }
  };
  auto stage_wait = [&]() {
    if (w == 0) __builtin_amdgcn_s_wait_tensorcnt(0);
  };
#else
  const int arow = tid >> 1;            // 0..127
  const int ac   = (tid & 1) * 16;      // half offset 0/16
  auto stage = [&](int kt, int buf) {
    // A tile: 128 rows x 32 halves; 2x16B per thread, contiguous 32B.
    const _Float16* Ap = A + (size_t)(bm + arow) * K + kt * 32 + ac;
    cp16(Ap,     &sA[buf][arow][ac]);
    cp16(Ap + 8, &sA[buf][arow][ac + 8]);
    // B tile: 256 rows x 32 halves; 4x16B per thread, contiguous 64B.
    const _Float16* Bp = BT + (size_t)(bn + tid) * K + kt * 32;
#pragma unroll
    for (int i = 0; i < 4; i++) cp16(Bp + 8 * i, &sB[buf][tid][8 * i]);
    if (kt + 1 < nk) {
      __builtin_prefetch(Ap + 32, 0, 1);
      __builtin_prefetch(Bp + 32, 0, 1);
    }
  };
  auto stage_wait = [&]() { async_wait0(); };
#endif

  stage(0, 0);
  stage_wait();
  __syncthreads();

  for (int kt = 0; kt < nk; ++kt) {
    const int buf = kt & 1;
    if (kt + 1 < nk) stage(kt + 1, buf ^ 1);

    V16 af[4], bf[4];
#pragma unroll
    for (int mt = 0; mt < 4; ++mt) {
      const int r = wrow + mt * 16 + lrow;
      af[mt].h[0] = *(const v8h*)&sA[buf][r][lhi * 8];
      af[mt].h[1] = *(const v8h*)&sA[buf][r][16 + lhi * 8];
    }
#pragma unroll
    for (int nt = 0; nt < 4; ++nt) {
      const int n = wcol + nt * 16 + lrow;
      bf[nt].h[0] = *(const v8h*)&sB[buf][n][lhi * 8];
      bf[nt].h[1] = *(const v8h*)&sB[buf][n][16 + lhi * 8];
    }
#pragma unroll
    for (int mt = 0; mt < 4; ++mt)
#pragma unroll
      for (int nt = 0; nt < 4; ++nt)
        acc[mt][nt] = wmma16(af[mt].v, bf[nt].v, acc[mt][nt]);

    stage_wait();
    __syncthreads();
  }

  // Epilogue (C layout: lane -> col lrow, VGPR r -> row r + 8*lhi).
#pragma unroll
  for (int mt = 0; mt < 4; ++mt) {
#pragma unroll
    for (int nt = 0; nt < 4; ++nt) {
      const int n = bn + wcol + nt * 16 + lrow;
      const float bv = bias ? bias[n] : 0.0f;
#pragma unroll
      for (int r = 0; r < 8; ++r) {
        const int m = bm + wrow + mt * 16 + r + 8 * lhi;
        float v = acc[mt][nt][r] + bv;
        if (RES)  v += res[(size_t)m * N + n];
        if (GELU) v = gelu_f(v);
        if (OUTH) Ch[(size_t)m * N + n] = (_Float16)v;
        else      Cf[(size_t)m * N + n] = v;
      }
    }
  }
}

// ---------------------------------------------------------------------------
// Flash attention, S=1024, HD=64; f16 q/k/v (head-interleaved rows, stride
// rowStride), 1/8 scale applied to scores. One block per (b,h,64 q-rows),
// 128 threads = 4 waves x 16 q-rows. Output f16 (OUTH) or f32.
// ---------------------------------------------------------------------------
template <bool OUTH>
__global__ __launch_bounds__(128)
void attn_h(const _Float16* __restrict__ qB, const _Float16* __restrict__ kB,
            const _Float16* __restrict__ vB, float* __restrict__ outF,
            _Float16* __restrict__ outH, int rowStride) {
  __shared__ _Float16 sQ[64][64];       // [qrow][hd]
  __shared__ _Float16 sK[64][64];       // [key][hd] (B-transposed layout)
  __shared__ _Float16 sVt[64][64];      // [hd][key]
  __shared__ _Float16 sP[4][16][64];    // per-wave probs [qrow][key]

  const int tid  = threadIdx.x;
  const int lane = tid & 31;
  const int lrow = lane & 15;
  const int lhi  = lane >> 4;
  const int w    = tid >> 5;

  const int blk = blockIdx.x;
  const int qt  = blk & 15;
  const int h   = (blk >> 4) & 15;
  const int b   = blk >> 8;

  const size_t headOff = (size_t)h * 64;
  const size_t rowBase = (size_t)b * 1024;

  {  // stage Q once (raw f16 copy; 4x16B per thread)
    const int r = tid >> 1;
    const int c = (tid & 1) * 32;
    const _Float16* Qp =
        qB + (rowBase + qt * 64 + r) * (size_t)rowStride + headOff + c;
#pragma unroll
    for (int i = 0; i < 4; i++) cp16(Qp + 8 * i, &sQ[r][c + 8 * i]);
  }
  async_wait0();
  __syncthreads();

  V16 qa[2];
#pragma unroll
  for (int ks = 0; ks < 2; ++ks) {
    const int r = w * 16 + lrow;
    qa[ks].h[0] = *(const v8h*)&sQ[r][ks * 32 + lhi * 8];
    qa[ks].h[1] = *(const v8h*)&sQ[r][ks * 32 + 16 + lhi * 8];
  }

  float m_run[8], s_run[8];
#pragma unroll
  for (int r = 0; r < 8; r++) { m_run[r] = -1e30f; s_run[r] = 0.0f; }
  const v8f vzero = {};
  v8f O[4];
#pragma unroll
  for (int nt = 0; nt < 4; nt++) O[nt] = vzero;

  for (int j = 0; j < 16; ++j) {        // 16 key blocks of 64
    const int jb = j * 64;
    __syncthreads();                    // prior readers of sK/sVt done
    {                                   // stage K (raw copy, async)
      const int r = tid >> 1;
      const int c = (tid & 1) * 32;
      const _Float16* Kp =
          kB + (rowBase + jb + r) * (size_t)rowStride + headOff + c;
#pragma unroll
      for (int i = 0; i < 4; i++) cp16(Kp + 8 * i, &sK[r][c + 8 * i]);
    }
    {                                   // stage V transposed (manual)
      const int key = tid >> 1;
      const int h0  = (tid & 1) * 32;
      const _Float16* Vp =
          vB + (rowBase + jb + key) * (size_t)rowStride + headOff + h0;
#pragma unroll
      for (int i = 0; i < 4; i++) {
        v8h vv = ((const v8h*)Vp)[i];
#pragma unroll
        for (int c = 0; c < 8; c++) sVt[h0 + i * 8 + c][key] = vv[c];
      }
    }
    async_wait0();
    __syncthreads();

    // Scores: 16 q-rows x 64 keys = 4 tiles x 2 K-steps.
    v8f sacc[4];
#pragma unroll
    for (int nt = 0; nt < 4; nt++) sacc[nt] = vzero;
#pragma unroll
    for (int ks = 0; ks < 2; ++ks) {
      V16 kf[4];
#pragma unroll
      for (int nt = 0; nt < 4; ++nt) {
        const int key = nt * 16 + lrow;
        kf[nt].h[0] = *(const v8h*)&sK[key][ks * 32 + lhi * 8];
        kf[nt].h[1] = *(const v8h*)&sK[key][ks * 32 + 16 + lhi * 8];
      }
#pragma unroll
      for (int nt = 0; nt < 4; ++nt)
        sacc[nt] = wmma16(qa[ks].v, kf[nt].v, sacc[nt]);
    }
#pragma unroll
    for (int nt = 0; nt < 4; nt++)
#pragma unroll
      for (int r = 0; r < 8; r++) sacc[nt][r] *= 0.125f;  // 1/sqrt(HD)

    // Online softmax; lane holds rows (r + 8*lhi).
    float mnew[8], rsum[8];
#pragma unroll
    for (int r = 0; r < 8; r++) {
      float v = sacc[0][r];
#pragma unroll
      for (int nt = 1; nt < 4; nt++) v = fmaxf(v, sacc[nt][r]);
#pragma unroll
      for (int msk = 8; msk >= 1; msk >>= 1) v = fmaxf(v, __shfl_xor(v, msk, 32));
      mnew[r] = fmaxf(m_run[r], v);
    }
#pragma unroll
    for (int r = 0; r < 8; r++) {
      float s = 0.0f;
#pragma unroll
      for (int nt = 0; nt < 4; nt++) {
        const float e = __expf(sacc[nt][r] - mnew[r]);
        sacc[nt][r] = e;
        s += e;
      }
#pragma unroll
      for (int msk = 8; msk >= 1; msk >>= 1) s += __shfl_xor(s, msk, 32);
      rsum[r] = s;
    }
#pragma unroll
    for (int r = 0; r < 8; r++) {
      const float alpha = __expf(m_run[r] - mnew[r]);
      s_run[r] = s_run[r] * alpha + rsum[r];
      m_run[r] = mnew[r];
#pragma unroll
      for (int nt = 0; nt < 4; nt++) O[nt][r] *= alpha;
    }

    // Spill P (f16) and reload as A fragments.
#pragma unroll
    for (int nt = 0; nt < 4; nt++)
#pragma unroll
      for (int r = 0; r < 8; r++)
        sP[w][r + 8 * lhi][nt * 16 + lrow] = (_Float16)sacc[nt][r];
    __syncthreads();

#pragma unroll
    for (int ks = 0; ks < 2; ++ks) {
      V16 pa;
      pa.h[0] = *(const v8h*)&sP[w][lrow][ks * 32 + lhi * 8];
      pa.h[1] = *(const v8h*)&sP[w][lrow][ks * 32 + 16 + lhi * 8];
      V16 vf_[4];
#pragma unroll
      for (int nt = 0; nt < 4; ++nt) {
        const int hd = nt * 16 + lrow;
        vf_[nt].h[0] = *(const v8h*)&sVt[hd][ks * 32 + lhi * 8];
        vf_[nt].h[1] = *(const v8h*)&sVt[hd][ks * 32 + 16 + lhi * 8];
      }
#pragma unroll
      for (int nt = 0; nt < 4; ++nt)
        O[nt] = wmma16(pa.v, vf_[nt].v, O[nt]);
    }
  }

  // Normalize and write merged heads: out[b,s,h*64+hd], stride D=1024.
#pragma unroll
  for (int r = 0; r < 8; r++) {
    const float inv = 1.0f / s_run[r];
    const size_t row = rowBase + (size_t)qt * 64 + w * 16 + r + 8 * lhi;
#pragma unroll
    for (int nt = 0; nt < 4; nt++) {
      const size_t idx = row * 1024 + h * 64 + nt * 16 + lrow;
      if (OUTH) outH[idx] = (_Float16)(O[nt][r] * inv);
      else      outF[idx] = O[nt][r] * inv;
    }
  }
}

// ---------------------------------------------------------------------------
// Host-side orchestration.
// ---------------------------------------------------------------------------
extern "C" void kernel_launch(void* const* d_in, const int* in_sizes, int n_in,
                              void* d_out, int out_size, void* d_ws, size_t ws_size,
                              hipStream_t stream) {
  (void)in_sizes; (void)n_in; (void)out_size; (void)ws_size;

  const int*   tokens = (const int*)d_in[0];
  const float* emb    = (const float*)d_in[1];
  const float* lm_w   = (const float*)d_in[2];
  const float* ln1g   = (const float*)d_in[3];
  const float* ln1b   = (const float*)d_in[4];
  const float* qkvw   = (const float*)d_in[5];
  const float* qkvbs  = (const float*)d_in[6];
  const float* outw   = (const float*)d_in[7];
  const float* outb   = (const float*)d_in[8];
  const float* ln2g   = (const float*)d_in[9];
  const float* ln2b   = (const float*)d_in[10];
  const float* ff1w   = (const float*)d_in[11];
  const float* ff1b   = (const float*)d_in[12];
  const float* ff2w   = (const float*)d_in[13];
  const float* ff2b   = (const float*)d_in[14];

  const size_t BS = 2048;               // B*S
  float* x2 = (float*)d_ws;             // [BS,1024] f32 residual stream (8MB)
  _Float16* hp = (_Float16*)(x2 + BS * 1024);
  _Float16* xh    = hp;                       hp += BS * 1024;   // emb f16
  _Float16* hh    = hp;                       hp += BS * 1024;   // LN out
  _Float16* qkvh  = hp;                       hp += BS * 3072;
  _Float16* ah    = hp;                       hp += BS * 1024;   // attn out
  _Float16* uh    = hp;                       hp += BS * 4096;   // FF mid
  _Float16* x2h   = hp;                       hp += BS * 1024;   // x2 as f16
  _Float16* wqkvT = hp;                       hp += (size_t)4 * 3072 * 1024;
  _Float16* woutT = hp;                       hp += (size_t)4 * 1024 * 1024;
  _Float16* wff1T = hp;                       hp += (size_t)4 * 4096 * 1024;
  _Float16* wff2T = hp;                       hp += (size_t)4 * 1024 * 4096;
  _Float16* wlmT  = hp;                       hp += (size_t)32000 * 1024;
  // total ws: 8MB f32 + ~212MB f16

  // Pre-transpose+convert all weights to f16 [N,K].
  for (int l = 0; l < 4; ++l) {
    convT_k<<<dim3(48, 16), 256, 0, stream>>>(
        qkvw + (size_t)l * 1024 * 3072, wqkvT + (size_t)l * 3072 * 1024, 1024, 3072);
    convT_k<<<dim3(16, 16), 256, 0, stream>>>(
        outw + (size_t)l * 1024 * 1024, woutT + (size_t)l * 1024 * 1024, 1024, 1024);
    convT_k<<<dim3(64, 16), 256, 0, stream>>>(
        ff1w + (size_t)l * 1024 * 4096, wff1T + (size_t)l * 4096 * 1024, 1024, 4096);
    convT_k<<<dim3(16, 64), 256, 0, stream>>>(
        ff2w + (size_t)l * 4096 * 1024, wff2T + (size_t)l * 1024 * 4096, 4096, 1024);
  }
  convT_k<<<dim3(500, 16), 256, 0, stream>>>(lm_w, wlmT, 1024, 32000);

  embed_k<<<2048, 256, 0, stream>>>(tokens, emb, xh);

  // Initial attention: q=k=v = embeddings; becomes the f32 residual base.
  attn_h<false><<<512, 128, 0, stream>>>(xh, xh, xh, x2, nullptr, 1024);

  for (int l = 0; l < 4; ++l) {
    ln_k<<<2048, 256, 0, stream>>>(x2, ln1g + l * 1024, ln1b + l * 1024, hh);
    gemm_h<false, false, true><<<dim3(12, 16), 256, 0, stream>>>(
        hh, wqkvT + (size_t)l * 3072 * 1024, qkvbs + l * 3072, nullptr,
        nullptr, qkvh, 2048, 3072, 1024);
    attn_h<true><<<512, 128, 0, stream>>>(qkvh, qkvh + 1024, qkvh + 2048,
                                          nullptr, ah, 3072);
    gemm_h<false, true, false><<<dim3(4, 16), 256, 0, stream>>>(
        ah, woutT + (size_t)l * 1024 * 1024, outb + l * 1024, x2,
        x2, nullptr, 2048, 1024, 1024);
    ln_k<<<2048, 256, 0, stream>>>(x2, ln2g + l * 1024, ln2b + l * 1024, hh);
    gemm_h<true, false, true><<<dim3(16, 16), 256, 0, stream>>>(
        hh, wff1T + (size_t)l * 4096 * 1024, ff1b + l * 4096, nullptr,
        nullptr, uh, 2048, 4096, 1024);
    gemm_h<false, true, false><<<dim3(4, 16), 256, 0, stream>>>(
        uh, wff2T + (size_t)l * 1024 * 4096, ff2b + l * 1024, x2,
        x2, nullptr, 2048, 1024, 4096);
  }

  // LM head: x2 -> f16, then [2048,1024] @ [1024,32000] -> f32 logits.
  cast_h_k<<<2048, 256, 0, stream>>>(x2, x2h);
  gemm_h<false, false, false><<<dim3(125, 16), 256, 0, stream>>>(
      x2h, wlmT, nullptr, nullptr, (float*)d_out, nullptr, 2048, 32000, 1024);
}
